// RPR_Multihead_Attention_77618648973877
// MI455X (gfx1250) — compile-verified, matrix operand order checked by reading, not run on hardware
//
#include <hip/hip_runtime.h>
#include <hip/hip_bf16.h>
#include <math.h>

// Problem constants (from the reference)
#define B_    8
#define C_    512
#define T_    1024
#define H_    8
#define HD_   64
#define CLIP_ 4
#define COUT_ 512

typedef __attribute__((ext_vector_type(16))) __bf16        v16bf;
typedef __attribute__((ext_vector_type(8)))  float         v8f;
typedef __attribute__((ext_vector_type(4)))  unsigned int  u32x4;
typedef __attribute__((ext_vector_type(8)))  unsigned int  u32x8;

// ---------------------------------------------------------------------------
// WMMA bf16 fragment loader (CDNA5 v_wmma_f32_16x16x32_bf16 operand layout).
// A matrix 16x32 (MxK): lane<16 -> row=lane, K in {0..7,16..23};
//                       lane>=16 -> row=lane-16, K in {8..15,24..31}.
// B (KxN) mirrors this, so loading B^T row-major with the same function is
// correct (lane = column N). Two contiguous 16-byte runs per lane.
// ---------------------------------------------------------------------------
static __device__ inline v16bf ld_frag_bf16(const __bf16* tile, int ldk) {
  const int lane = threadIdx.x & 31;
  const int r  = lane & 15;
  const int k0 = (lane < 16) ? 0 : 8;
  const __bf16* p = tile + (size_t)r * ldk + k0;
  v16bf f;
#pragma unroll
  for (int e = 0; e < 8; ++e) f[e] = p[e];
#pragma unroll
  for (int e = 0; e < 8; ++e) f[8 + e] = p[16 + e];
  return f;
}

static __device__ inline v8f wmma_bf16(v16bf a, v16bf b, v8f c) {
  return __builtin_amdgcn_wmma_f32_16x16x32_bf16(
      /*neg_a=*/false, a, /*neg_b=*/false, b,
      /*c_mod=*/(short)0, c, /*reuse_a=*/false, /*reuse_b=*/false);
}

// ---------------------------------------------------------------------------
// Kernel 0a: f32 -> bf16 weight conversion
// ---------------------------------------------------------------------------
__global__ void cvt_f32_bf16(const float* __restrict__ src,
                             __bf16* __restrict__ dst, int n) {
  int i = blockIdx.x * blockDim.x + threadIdx.x;
  if (i < n) dst[i] = (__bf16)src[i];
}

// Kernel 0b: queries [B,C,T] f32 -> x16 [B,T,C] bf16 (transpose for B-operand)
__global__ void xpose_queries(const float* __restrict__ q,
                              __bf16* __restrict__ x16) {
  size_t i = blockIdx.x * (size_t)blockDim.x + threadIdx.x;
  if (i >= (size_t)B_ * C_ * T_) return;
  int t = (int)(i % T_);
  size_t bc = i / T_;
  int c = (int)(bc % C_);
  int b = (int)(bc / C_);
  x16[((size_t)b * T_ + t) * C_ + c] = (__bf16)q[i];
}

// ---------------------------------------------------------------------------
// Kernel 1: QKV projection GEMM.  D[o, n] = sum_c W[o,c] * x16[n,c] + bias[o]
// One wave computes a 32x32 output block (2x2 WMMA tiles): 4 fragment loads
// feed 4 WMMAs per K-step (2x the FLOP/byte of a single-tile wave).
// mode 0/1 -> store as [b,h,t,d] (q16/k16); mode 2 -> store as [b,h,d,t] (vT)
// ---------------------------------------------------------------------------
__global__ void qkv_gemm(const __bf16* __restrict__ W16,
                         const float*  __restrict__ bias,
                         const __bf16* __restrict__ x16,
                         __bf16* __restrict__ out, int mode) {
  int wave = (int)((blockIdx.x * (size_t)blockDim.x + threadIdx.x) >> 5);
  int op = wave & 15;          // 16 o-pairs (32 channels each)
  int np = wave >> 4;          // 256 n-pairs (32 tokens each)
  if (np >= (B_ * T_) / 32) return;

  const __bf16* A0 = W16 + (size_t)(op * 32) * C_;
  const __bf16* A1 = A0 + (size_t)16 * C_;
  const __bf16* B0 = x16 + (size_t)(np * 32) * C_;
  const __bf16* B1 = B0 + (size_t)16 * C_;
  v8f acc[2][2] = {};
  for (int k = 0; k < C_; k += 32) {
    __builtin_prefetch(A0 + k + 64, 0, 0);
    __builtin_prefetch(B0 + k + 64, 0, 0);
    v16bf a0 = ld_frag_bf16(A0 + k, C_);
    v16bf a1 = ld_frag_bf16(A1 + k, C_);
    v16bf b0 = ld_frag_bf16(B0 + k, C_);
    v16bf b1 = ld_frag_bf16(B1 + k, C_);
    acc[0][0] = wmma_bf16(a0, b0, acc[0][0]);
    acc[0][1] = wmma_bf16(a0, b1, acc[0][1]);
    acc[1][0] = wmma_bf16(a1, b0, acc[1][0]);
    acc[1][1] = wmma_bf16(a1, b1, acc[1][1]);
  }
  const int lane = threadIdx.x & 31;
#pragma unroll
  for (int pi = 0; pi < 2; ++pi) {
#pragma unroll
    for (int pj = 0; pj < 2; ++pj) {
      int n = np * 32 + pj * 16 + (lane & 15);
      int bb = n / T_, t = n % T_;
#pragma unroll
      for (int i = 0; i < 8; ++i) {
        int o = op * 32 + pi * 16 + i + ((lane < 16) ? 0 : 8);
        float v = acc[pi][pj][i] + bias[o];
        int h = o / HD_, d = o % HD_;
        size_t idx = (mode == 2)
            ? (((size_t)bb * H_ + h) * HD_ + d) * T_ + t     // v transposed
            : (((size_t)bb * H_ + h) * T_  + t) * HD_ + d;   // q / k
        out[idx] = (__bf16)v;
      }
    }
  }
}

// ---------------------------------------------------------------------------
// Kernel 2: qwK[bh,t,j] = q[bh,t,:] . wK[j,:]   (j = 0..8, band bias terms)
// ---------------------------------------------------------------------------
__global__ void qwk_kernel(const __bf16* __restrict__ q16,
                           const float*  __restrict__ wK,
                           float* __restrict__ qwK) {
  int i = blockIdx.x * blockDim.x + threadIdx.x;   // over B*H*T rows
  if (i >= B_ * H_ * T_) return;
  const __bf16* qr = q16 + (size_t)i * HD_;
  float qv[HD_];
#pragma unroll
  for (int d = 0; d < HD_; ++d) qv[d] = (float)qr[d];
#pragma unroll
  for (int j = 0; j < 2 * CLIP_ + 1; ++j) {
    float s = 0.f;
#pragma unroll
    for (int d = 0; d < HD_; ++d) s += qv[d] * wK[j * HD_ + d];
    qwK[(size_t)i * (2 * CLIP_ + 1) + j] = s;
  }
}

// ---------------------------------------------------------------------------
// Kernel 3: fused scores + banded rel-K bias + softmax.
// One workgroup (4 waves) handles (b,h, 64-row t-block). The 64x1024 f32
// score slab lives in LDS (256 KB of the WGP's 320 KB).  k-tiles are staged
// into LDS via the Tensor Data Mover (tensor_load_to_lds, TENSORcnt), then
// consumed as WMMA B-fragments via ds loads.
// ---------------------------------------------------------------------------
#define KBUF_LDS_OFF ((unsigned)((64 * T_ + 128) * sizeof(float)))

__global__ void scores_softmax(const __bf16* __restrict__ q16,
                               const __bf16* __restrict__ k16,
                               const float*  __restrict__ qwK,
                               float*  __restrict__ align_out,
                               __bf16* __restrict__ P16) {
  extern __shared__ float lds[];
  float*  S    = lds;                       // [64][1024]
  float*  rmax = lds + 64 * T_;             // [64]
  float*  rinv = rmax + 64;                 // [64]
  __bf16* kbuf = (__bf16*)(rinv + 64);      // [64][HD_] staged k rows

  const int bh   = blockIdx.x / (T_ / 64);
  const int tblk = blockIdx.x % (T_ / 64);
  const int wave = threadIdx.x >> 5;
  const int lane = threadIdx.x & 31;
  const int tt   = tblk * 4 + wave;         // this wave's t-tile (16 rows)

  // A fragments (q rows) reused across all s-tiles: load once from global.
  const __bf16* Abase = q16 + ((size_t)bh * T_ + tt * 16) * HD_;
  v16bf a0 = ld_frag_bf16(Abase, HD_);
  v16bf a1 = ld_frag_bf16(Abase + 32, HD_);

  for (int sblk = 0; sblk < T_ / 64; ++sblk) {
    if (wave == 0) {
      // Build the Tensor DMA Descriptor (D#) for a 64x64 bf16 tile copy.
      unsigned long long ga =
          (unsigned long long)(k16 + ((size_t)bh * T_ + sblk * 64) * HD_);
      u32x4 g0;
      g0[0] = 1u;                                   // count=1 (valid D#)
      g0[1] = KBUF_LDS_OFF;                         // lds_addr (bytes)
      g0[2] = (unsigned)ga;                         // global_addr[31:0]
      g0[3] = (unsigned)((ga >> 32) & 0x01FFFFFFu)  // global_addr[56:32]
              | (2u << 30);                         // type=2 ("image")
      u32x8 g1;
      const unsigned dim0 = HD_, dim1 = 64, tile0 = HD_, tile1 = 64;
      const unsigned long long st0 = HD_;           // row stride (elements)
      g1[0] = (1u << 16);                           // data_size=1 -> 2 bytes
      g1[1] = (dim0 & 0xFFFFu) << 16;               // tensor_dim0[15:0]
      g1[2] = ((dim0 >> 16) & 0xFFFFu) | ((dim1 & 0xFFFFu) << 16);
      g1[3] = ((dim1 >> 16) & 0xFFFFu) | ((tile0 & 0xFFFFu) << 16);
      g1[4] = (tile1 & 0xFFFFu);                    // tile_dim1, tile_dim2=0
      g1[5] = (unsigned)st0;                        // tensor_dim0_stride[31:0]
      g1[6] = (unsigned)((st0 >> 32) & 0xFFFFu);    // stride[47:32]
      g1[7] = 0u;
      asm volatile("tensor_load_to_lds %0, %1" :: "s"(g0), "s"(g1) : "memory");
      __builtin_amdgcn_s_wait_tensorcnt(0);
    }
    __syncthreads();                                // kbuf ready for all waves

#pragma unroll
    for (int st = 0; st < 4; ++st) {
      const __bf16* Bbase = kbuf + (st * 16) * HD_;
      v16bf b0 = ld_frag_bf16(Bbase, HD_);          // ds loads
      v16bf b1 = ld_frag_bf16(Bbase + 32, HD_);
      v8f acc = {};
      acc = wmma_bf16(a0, b0, acc);
      acc = wmma_bf16(a1, b1, acc);
      int s = sblk * 64 + st * 16 + (lane & 15);
#pragma unroll
      for (int i = 0; i < 8; ++i) {
        int rl = wave * 16 + i + ((lane < 16) ? 0 : 8);
        S[rl * T_ + s] = acc[i];
      }
    }
    __syncthreads();                                // done with kbuf
  }

  const float scale = 0.125f;                       // 1/sqrt(HD)
  if (threadIdx.x < 64) {
    int r = threadIdx.x;
    int t = tblk * 64 + r;
    const float* qw = qwK + ((size_t)bh * T_ + t) * (2 * CLIP_ + 1);
#pragma unroll
    for (int j = 0; j < 2 * CLIP_ + 1; ++j) {       // banded rel-key bias
      int s = t + j - CLIP_;
      if (s >= 0 && s < T_) S[r * T_ + s] += qw[j];
    }
    float m = -3.0e38f;
    for (int s = 0; s < T_; ++s) m = fmaxf(m, S[r * T_ + s]);
    float sum = 0.f;
    for (int s = 0; s < T_; ++s) sum += __expf(scale * (S[r * T_ + s] - m));
    rmax[r] = m;
    rinv[r] = 1.0f / sum;
  }
  __syncthreads();

  const size_t base = ((size_t)bh * T_ + tblk * 64) * T_;
  for (int idx = threadIdx.x; idx < 64 * T_; idx += blockDim.x) {
    int r = idx >> 10;
    float p = __expf(scale * (S[idx] - rmax[r])) * rinv[r];
    align_out[base + idx] = p;                      // f32 alignments output
    P16[base + idx] = (__bf16)p;                    // bf16 copy for AV GEMM
  }
}

// ---------------------------------------------------------------------------
// Kernel 4: O[t,d] = sum_s P[t,s] v[s,d]  (+ banded rel-V bias), per (b,h).
// One wave computes 32(t) x 32(d) via 2x2 WMMA tiles; B operand is vT so
// loads are contiguous in K=s.  attn stored bf16 as [b, t, c] (c = h*HD+d).
// ---------------------------------------------------------------------------
__global__ void av_gemm(const __bf16* __restrict__ P16,
                        const __bf16* __restrict__ v16T,
                        const float*  __restrict__ align,
                        const float*  __restrict__ wV,
                        __bf16* __restrict__ attn16) {
  int wave = (int)((blockIdx.x * (size_t)blockDim.x + threadIdx.x) >> 5);
  int bh = wave >> 6;                 // 64 blocks per (b,h)
  int tl = wave & 63;
  int tp = tl >> 1, dp = tl & 1;      // 32 t-pairs x 2 d-pairs
  if (bh >= B_ * H_) return;

  const __bf16* A0 = P16  + ((size_t)bh * T_  + tp * 32) * T_;
  const __bf16* A1 = A0 + (size_t)16 * T_;
  const __bf16* B0 = v16T + ((size_t)bh * HD_ + dp * 32) * T_;
  const __bf16* B1 = B0 + (size_t)16 * T_;
  v8f acc[2][2] = {};
  for (int k = 0; k < T_; k += 32) {
    __builtin_prefetch(A0 + k + 64, 0, 0);
    __builtin_prefetch(B0 + k + 64, 0, 0);
    v16bf a0 = ld_frag_bf16(A0 + k, T_);
    v16bf a1 = ld_frag_bf16(A1 + k, T_);
    v16bf b0 = ld_frag_bf16(B0 + k, T_);
    v16bf b1 = ld_frag_bf16(B1 + k, T_);
    acc[0][0] = wmma_bf16(a0, b0, acc[0][0]);
    acc[0][1] = wmma_bf16(a0, b1, acc[0][1]);
    acc[1][0] = wmma_bf16(a1, b0, acc[1][0]);
    acc[1][1] = wmma_bf16(a1, b1, acc[1][1]);
  }
  const int lane = threadIdx.x & 31;
  const int bb = bh / H_, h = bh % H_;
#pragma unroll
  for (int pj = 0; pj < 2; ++pj) {
    int d = dp * 32 + pj * 16 + (lane & 15);
    float wv[2 * CLIP_ + 1];
#pragma unroll
    for (int j = 0; j < 2 * CLIP_ + 1; ++j) wv[j] = wV[j * HD_ + d];
#pragma unroll
    for (int pi = 0; pi < 2; ++pi) {
#pragma unroll
      for (int i = 0; i < 8; ++i) {
        int t = tp * 32 + pi * 16 + i + ((lane < 16) ? 0 : 8);
        float v = acc[pi][pj][i];
        const float* Prow = align + ((size_t)bh * T_ + t) * T_;
#pragma unroll
        for (int j = 0; j < 2 * CLIP_ + 1; ++j) {   // banded rel-value bias
          int s = t + j - CLIP_;
          if (s >= 0 && s < T_) v += Prow[s] * wv[j];
        }
        attn16[((size_t)bb * T_ + t) * C_ + h * HD_ + d] = (__bf16)v;
      }
    }
  }
}

// ---------------------------------------------------------------------------
// Kernel 5: out[b,o,t] = sum_c Wp[o,c] attn[b,t,c] + bp[o]   (2x2 tiles)
// ---------------------------------------------------------------------------
__global__ void out_proj(const __bf16* __restrict__ Wp16,
                         const float*  __restrict__ bp,
                         const __bf16* __restrict__ attn16,
                         float* __restrict__ out) {
  int wave = (int)((blockIdx.x * (size_t)blockDim.x + threadIdx.x) >> 5);
  int bb = wave >> 9;                  // 512 blocks per batch
  int tl = wave & 511;
  int op = tl >> 5, tp = tl & 31;      // 16 o-pairs x 32 t-pairs
  if (bb >= B_) return;

  const __bf16* A0 = Wp16   + (size_t)(op * 32) * C_;
  const __bf16* A1 = A0 + (size_t)16 * C_;
  const __bf16* B0 = attn16 + ((size_t)bb * T_ + tp * 32) * C_;
  const __bf16* B1 = B0 + (size_t)16 * C_;
  v8f acc[2][2] = {};
  for (int k = 0; k < C_; k += 32) {
    __builtin_prefetch(A0 + k + 64, 0, 0);
    __builtin_prefetch(B0 + k + 64, 0, 0);
    v16bf a0 = ld_frag_bf16(A0 + k, C_);
    v16bf a1 = ld_frag_bf16(A1 + k, C_);
    v16bf b0 = ld_frag_bf16(B0 + k, C_);
    v16bf b1 = ld_frag_bf16(B1 + k, C_);
    acc[0][0] = wmma_bf16(a0, b0, acc[0][0]);
    acc[0][1] = wmma_bf16(a0, b1, acc[0][1]);
    acc[1][0] = wmma_bf16(a1, b0, acc[1][0]);
    acc[1][1] = wmma_bf16(a1, b1, acc[1][1]);
  }
  const int lane = threadIdx.x & 31;
#pragma unroll
  for (int pi = 0; pi < 2; ++pi) {
#pragma unroll
    for (int pj = 0; pj < 2; ++pj) {
      int t = tp * 32 + pj * 16 + (lane & 15);
#pragma unroll
      for (int i = 0; i < 8; ++i) {
        int o = op * 32 + pi * 16 + i + ((lane < 16) ? 0 : 8);
        out[((size_t)bb * COUT_ + o) * T_ + t] = acc[pi][pj][i] + bp[o];
      }
    }
  }
}

// ---------------------------------------------------------------------------
extern "C" void kernel_launch(void* const* d_in, const int* in_sizes, int n_in,
                              void* d_out, int out_size, void* d_ws, size_t ws_size,
                              hipStream_t stream) {
  (void)in_sizes; (void)n_in; (void)out_size; (void)ws_size;

  const float* queries = (const float*)d_in[0];
  const float* Wq = (const float*)d_in[1];
  const float* bq = (const float*)d_in[2];
  const float* Wk = (const float*)d_in[3];
  const float* bk = (const float*)d_in[4];
  const float* Wv = (const float*)d_in[5];
  const float* bv = (const float*)d_in[6];
  const float* Wp = (const float*)d_in[7];
  const float* bp = (const float*)d_in[8];
  const float* wK = (const float*)d_in[9];    // [1, 9, 64]
  const float* wV = (const float*)d_in[10];   // [1, 9, 64]

  float* out_proj_dst = (float*)d_out;                               // [B,COUT,T]
  float* align_dst    = (float*)d_out + (size_t)B_ * COUT_ * T_;     // [B,H,T,T]

  // Workspace carve-up
  char* ws = (char*)d_ws;
  const size_t nBTC = (size_t)B_ * T_ * C_;        // 4,194,304
  const size_t nW   = (size_t)C_ * C_;             // 262,144
  __bf16* x16    = (__bf16*)(ws);                               // [b,t,c]
  __bf16* Wq16   = (__bf16*)(ws + nBTC * 2);
  __bf16* Wk16   = Wq16 + nW;
  __bf16* Wv16   = Wk16 + nW;
  __bf16* Wp16   = Wv16 + nW;
  __bf16* q16    = Wp16 + nW;                                    // [bh,t,d]
  __bf16* k16    = q16 + nBTC;                                   // [bh,t,d]
  __bf16* v16T   = k16 + nBTC;                                   // [bh,d,t]
  __bf16* attn16 = v16T + nBTC;                                  // [b,t,c]
  float*  qwK    = (float*)(attn16 + nBTC);                      // [bh,t,9]
  __bf16* P16    = (__bf16*)((char*)qwK + (size_t)B_*H_*T_*9*4); // [bh,t,s]

  // (0) precision conversion + transpose
  cvt_f32_bf16<<<(int)((nW + 255) / 256), 256, 0, stream>>>(Wq, Wq16, (int)nW);
  cvt_f32_bf16<<<(int)((nW + 255) / 256), 256, 0, stream>>>(Wk, Wk16, (int)nW);
  cvt_f32_bf16<<<(int)((nW + 255) / 256), 256, 0, stream>>>(Wv, Wv16, (int)nW);
  cvt_f32_bf16<<<(int)((nW + 255) / 256), 256, 0, stream>>>(Wp, Wp16, (int)nW);
  xpose_queries<<<(int)((nBTC + 255) / 256), 256, 0, stream>>>(queries, x16);

  // (1) QKV projections: 4096 waves (32x32 block each), 8 waves/block
  qkv_gemm<<<512, 256, 0, stream>>>(Wq16, bq, x16, q16, 0);
  qkv_gemm<<<512, 256, 0, stream>>>(Wk16, bk, x16, k16, 1);
  qkv_gemm<<<512, 256, 0, stream>>>(Wv16, bv, x16, v16T, 2);

  // (2) banded rel-key dot products
  qwk_kernel<<<(B_ * H_ * T_) / 256, 256, 0, stream>>>(q16, wK, qwK);

  // (3) fused scores + bias + softmax; 256.5 KB scores + 8 KB TDM k-buffer
  const size_t smem = (size_t)(64 * T_ + 128) * sizeof(float)
                    + (size_t)64 * HD_ * sizeof(__bf16);
  scores_softmax<<<B_ * H_ * (T_ / 64), 128, smem, stream>>>(
      q16, k16, qwK, align_dst, P16);

  // (4) attention * V + banded rel-value bias
  av_gemm<<<512, 256, 0, stream>>>(P16, v16T, align_dst, wV, attn16);

  // (5) output projection
  out_proj<<<512, 256, 0, stream>>>(Wp16, bp, attn16, out_proj_dst);
}
